// FDGRU_37031208026605
// MI455X (gfx1250) — compile-verified, matrix operand order checked by reading, not run on hardware
//
#include <hip/hip_runtime.h>
#include <math.h>

// ============================================================================
// FDGRU forward for MI455X (gfx1250, wave32, WMMA + TDM).
//
// d_in leaf order assumed (pytree insertion order of setup_inputs()):
//  0 inp_seq[1,4096,63] 1 h[1024] 2 h_start[1024] 3 bb_dims[3] 4 gt[1,4096,63]
//  5..10  inp_net  11..16 cmd_net  17..22 cube_net  23..28 dim_net
// 29..34 align_net 35..40 att_net  41..46 sym_net   47..52 squeeze_net
// 53..58 next_net  59..64 leaf_net
// 65 gru.W_ih 66 gru.W_hh 67 gru.b_ih 68 gru.b_hh   69 hier_ind (int)
//
// d_out: out[4096,63] | next_out[4096,1024] | leaf_out[4096,1] | h_new[1024]
// ============================================================================

#define S_LEN 4096
#define HDIM  1024
#define GRU_WGS 16

typedef __bf16 bf16_t;
typedef __attribute__((ext_vector_type(16))) __bf16 v16bf;
typedef __attribute__((ext_vector_type(8)))  __bf16 bf16x8;
typedef __attribute__((ext_vector_type(8)))  float  v8f;
typedef unsigned int v4u __attribute__((ext_vector_type(4)));
typedef int          v4i __attribute__((ext_vector_type(4)));
typedef int          v8i __attribute__((ext_vector_type(8)));

#if defined(__AMDGCN__) && __has_builtin(__builtin_amdgcn_tensor_load_to_lds)
#define HAS_TDM 1
#endif

// LDS B-tile pitch: 64 bf16 (128B) data + 16B TDM pad per row -> bank spread
#define BROW_PITCH 144

#if defined(HAS_TDM)
// ---------------------------------------------------------------------------
// Issue TDM: copy a [64 rows x 64 k] bf16 tile (row stride = K elements)
// from global to LDS with 16B padding after every 128B (pitch 144B).
// D# group0: count=1 | lds_addr<<32 ; global_addr | type=2.
// D# group1 bit layout per CDNA5 ISA 8.4.
// ---------------------------------------------------------------------------
__device__ __forceinline__ void tdm_load_tile_b(unsigned lds_addr,
                                                const bf16_t* gsrc,
                                                long long row_stride_elems) {
  union { unsigned long long q[2]; v4u v; } g0;
  g0.q[0] = 1ull | ((unsigned long long)lds_addr << 32);
  g0.q[1] = (((unsigned long long)(size_t)gsrc) & ((1ull << 57) - 1)) | (2ull << 62);

  union { unsigned long long q[4]; v8i v; } g1;
  const unsigned long long dsz = 1;            // 2-byte elements
  const unsigned long long tdim0 = 1u << 20;   // generous (tile always in range)
  const unsigned long long tdim1 = 1u << 20;
  const unsigned long long pad_en = 1, pad_iv = 4 /*32 dwords*/, pad_am = 3 /*4 dwords*/;
  const unsigned long long tile_k = 64, tile_n = 64;
  const unsigned long long s0 = (unsigned long long)row_stride_elems;
  g1.q[0] = (dsz << 16) | (pad_en << 20) | (pad_iv << 22) | (pad_am << 25)
          | ((tdim0 & 0xffffull) << 48);
  g1.q[1] = ((tdim0 >> 16) & 0xffffull) | ((tdim1 & 0xffffffffull) << 16)
          | (tile_k << 48);
  g1.q[2] = tile_n | ((s0 & 0xffffffffull) << 32);
  g1.q[3] = (s0 >> 32) & 0xffffull;            // dim1_stride unused (2D tile)

  v4i z4 = 0;
#if __clang_major__ >= 23
  v8i z8 = 0;
  __builtin_amdgcn_tensor_load_to_lds(g0.v, g1.v, z4, z4, z8, 0);
#else
  __builtin_amdgcn_tensor_load_to_lds(g0.v, g1.v, z4, z4, 0);
#endif
}
#endif

// ---------------------------------------------------------------------------
// Weight convert: W[K,N] f32 -> Wt[N,Kpad] bf16 (transposed, zero K-pad)
// ---------------------------------------------------------------------------
__global__ void k_convert_w(const float* __restrict__ W, bf16_t* __restrict__ Wt,
                            int K, int N, int Kpad) {
  long long idx = (long long)blockIdx.x * blockDim.x + threadIdx.x;
  long long tot = (long long)N * Kpad;
  if (idx >= tot) return;
  int n = (int)(idx / Kpad), k = (int)(idx % Kpad);
  float v = (k < K) ? W[(long long)k * N + n] : 0.f;
  Wt[idx] = (bf16_t)v;
}

// ---------------------------------------------------------------------------
// x0 = concat([inp_seq(63), bb(3), hier_onehot(3)]) padded 69 -> 128, bf16
// ---------------------------------------------------------------------------
__global__ void k_build_x0(const float* __restrict__ inp_seq,
                           const float* __restrict__ bb,
                           const int* __restrict__ hier,
                           bf16_t* __restrict__ dst, int S) {
  long long idx = (long long)blockIdx.x * blockDim.x + threadIdx.x;
  if (idx >= (long long)S * 128) return;
  int s = (int)(idx / 128), j = (int)(idx % 128);
  float v = 0.f;
  if (j < 63) {
    v = inp_seq[(long long)s * 63 + j];
  } else if (j < 66) {
    v = bb[j - 63];
  } else if (j < 69) {
    int hi = hier[0]; if (hi > 2) hi = 2; if (hi < 0) hi = 0;
    v = ((j - 66) == hi) ? 1.f : 0.f;
  }
  dst[idx] = (bf16_t)v;
}

// ---------------------------------------------------------------------------
// Row concat (bf16): dst[s] = [base(base_w) | extra(ew) | bcast(bw) | 0]
// ---------------------------------------------------------------------------
__global__ void k_concat(bf16_t* __restrict__ dst, int Kpad,
                         const bf16_t* __restrict__ base, int base_w,
                         const float* __restrict__ extra, int estride, int eoff, int ew,
                         const float* __restrict__ bcast, int bw, int S) {
  long long idx = (long long)blockIdx.x * blockDim.x + threadIdx.x;
  if (idx >= (long long)S * Kpad) return;
  int s = (int)(idx / Kpad), j = (int)(idx % Kpad);
  if (j < base_w) { dst[idx] = base[(long long)s * base_w + j]; return; }
  j -= base_w;
  float v = 0.f;
  if (ew > 0 && j < ew) {
    v = extra[(long long)s * estride + eoff + j];
  } else {
    if (ew > 0) j -= ew;
    if (bw > 0 && j < bw) v = bcast[j];
  }
  dst[idx] = (bf16_t)v;
}

// ---------------------------------------------------------------------------
// WMMA bf16 GEMM: C[M,N] = act(A[M,K] * Bt[N,K]^T + bias)
//   Block: 256 threads = 8 waves; wave w rows [m0,m0+32), block tile 256x64.
//   B tile (64 n x 64 k) staged in LDS once per block per K-chunk:
//     TDM pipeline (tensor_load_to_lds + s_wait_tensorcnt), double-buffered,
//     row pitch 144B (TDM pad) for bank spread. Fallback: manual staging.
//   Per k-step: batch-load 4 B frags from LDS + 2 A frags from global, then
//   issue 8 back-to-back WMMAs (loads clause under the XDL pipe).
//   A frag layout (ISA): lane L: row = L&15, half = L>>4,
//     elem e -> K = (e&7)+8*((e>>3)*2+half)  => two 16B loads per frag.
//   Requires M%256==0, N%64==0, K%64==0 (host pads K).
// ---------------------------------------------------------------------------
__device__ __forceinline__ v16bf load_frag(const bf16_t* __restrict__ rowp,
                                           int k0, int half) {
  union { v16bf v; bf16x8 h[2]; } u;
  u.h[0] = *(const bf16x8*)(rowp + k0 + 8 * half);
  u.h[1] = *(const bf16x8*)(rowp + k0 + 16 + 8 * half);
  return u.v;
}

__global__ __launch_bounds__(256) void k_gemm(
    const bf16_t* __restrict__ A, const bf16_t* __restrict__ Bt,
    const float* __restrict__ bias,
    float* __restrict__ Cf, bf16_t* __restrict__ Cb,
    int M, int N, int K, int act) {
  __shared__ char BsRaw[2][64 * BROW_PITCH];
  const int lane = threadIdx.x & 31;
  const int wave = threadIdx.x >> 5;
  const int half = lane >> 4;
  const int l16  = lane & 15;
  const int m0 = blockIdx.y * 256 + wave * 32;
  const int n0 = blockIdx.x * 64;

  v8f acc0[4] = {}, acc1[4] = {};
  const bf16_t* arow0 = A + (size_t)(m0 + l16) * K;
  const bf16_t* arow1 = A + (size_t)(m0 + 16 + l16) * K;
  const int nchunks = K >> 6;

#if defined(HAS_TDM)
  if (wave == 0)
    tdm_load_tile_b((unsigned)(size_t)&BsRaw[0][0], Bt + (size_t)n0 * K, K);
#endif

  for (int c = 0; c < nchunks; ++c) {
#if defined(HAS_TDM)
    const int buf = c & 1;
    if (wave == 0) {
      if (c + 1 < nchunks) {
        tdm_load_tile_b((unsigned)(size_t)&BsRaw[buf ^ 1][0],
                        Bt + (size_t)n0 * K + (c + 1) * 64, K);
        __builtin_amdgcn_s_wait_tensorcnt(1);   // chunk c DMA complete
      } else {
        __builtin_amdgcn_s_wait_tensorcnt(0);
      }
    }
    __syncthreads();                            // B tile c visible to all waves
#else
    const int buf = 0;
    __syncthreads();                            // previous chunk consumed
    for (int id = threadIdx.x; id < 512; id += 256) {
      int row = id >> 3, cb = id & 7;
      bf16x8 v = *(const bf16x8*)(Bt + (size_t)(n0 + row) * K + (c << 6) + (cb << 3));
      *(bf16x8*)&BsRaw[0][row * BROW_PITCH + cb * 16] = v;
    }
    __syncthreads();
#endif

    __builtin_prefetch((const void*)(arow0 + ((c + 2) << 6)), 0, 3);
    __builtin_prefetch((const void*)(arow1 + ((c + 2) << 6)), 0, 3);

#pragma unroll
    for (int s = 0; s < 2; ++s) {
      const int kk = (c << 6) + (s << 5);
      v16bf a0 = load_frag(arow0, kk, half);
      v16bf a1 = load_frag(arow1, kk, half);
      v16bf b[4];
#pragma unroll
      for (int nt = 0; nt < 4; ++nt) {          // batch B frags -> one clause
        const char* rp = &BsRaw[buf][(nt * 16 + l16) * BROW_PITCH];
        union { v16bf v; bf16x8 h[2]; } ub;
        ub.h[0] = *(const bf16x8*)(rp + s * 64 + 16 * half);
        ub.h[1] = *(const bf16x8*)(rp + s * 64 + 32 + 16 * half);
        b[nt] = ub.v;
      }
#pragma unroll
      for (int nt = 0; nt < 4; ++nt) {          // 8 back-to-back WMMAs
        acc0[nt] = __builtin_amdgcn_wmma_f32_16x16x32_bf16(
            false, a0, false, b[nt], (short)0, acc0[nt], false, false);
        acc1[nt] = __builtin_amdgcn_wmma_f32_16x16x32_bf16(
            false, a1, false, b[nt], (short)0, acc1[nt], false, false);
      }
    }
    __syncthreads();                            // protect LDS buffer reuse
  }

  // C layout: lane L, vgpr v -> row m0 + v + 8*(L>>4), col n0 + nt*16 + (L&15)
#pragma unroll
  for (int nt = 0; nt < 4; ++nt) {
    int col = n0 + nt * 16 + l16;
    float bv = bias ? bias[col] : 0.f;
#pragma unroll
    for (int v = 0; v < 8; ++v) {
      int row0 = m0 + v + 8 * half;
      int row1 = row0 + 16;
      float x0 = acc0[nt][v] + bv;
      float x1 = acc1[nt][v] + bv;
      if (act) {
        x0 = (x0 > 0.f) ? x0 : 0.2f * x0;       // leaky_relu slope 0.2
        x1 = (x1 > 0.f) ? x1 : 0.2f * x1;
      }
      if (Cf) {
        Cf[(size_t)row0 * N + col] = x0;
        Cf[(size_t)row1 * N + col] = x1;
      }
      if (Cb) {
        Cb[(size_t)row0 * N + col] = (bf16_t)x0;
        Cb[(size_t)row1 * N + col] = (bf16_t)x1;
      }
    }
  }
}

// ---------------------------------------------------------------------------
// Tiny final layers (N<=33): f32 matvec into strided output slice.
// ---------------------------------------------------------------------------
__global__ void k_head_out(const float* __restrict__ A, int K,
                           const float* __restrict__ W, const float* __restrict__ b,
                           float* __restrict__ outp, int ostride, int ooff,
                           int N, int S) {
  long long idx = (long long)blockIdx.x * blockDim.x + threadIdx.x;
  if (idx >= (long long)S * N) return;
  int s = (int)(idx / N), n = (int)(idx % N);
  float acc = b[n];
  const float* a = A + (size_t)s * K;
  for (int k = 0; k < K; ++k) acc = fmaf(a[k], W[(size_t)k * N + n], acc);
  outp[(size_t)s * ostride + ooff + n] = acc;
}

// ---------------------------------------------------------------------------
// GRU scan. 16 WGs x 1024 threads; WG b owns h[b*64 .. b*64+64).
// Thread (o,t16): 64-k partials for 3 gate columns (16B vector weight loads);
// 16-lane shuffle reduce. h exchanged via double-buffered global vector;
// per-step sync = cluster barrier (fast path when dispatched as a 16-WG
// cluster) + atomic arrive/spin (correct in any dispatch mode).
// Whht = W_hh transposed [3H,H] bf16 (~6MB, L2-resident). xw includes b_ih.
// ---------------------------------------------------------------------------
__global__ __launch_bounds__(1024) void k_gru_scan(
    const float* __restrict__ xw,        // [S, 3H]
    const bf16_t* __restrict__ Whht,     // [3H, H]
    const float* __restrict__ b_hh,      // [3H]
    float* __restrict__ hbuf,            // [2][H]
    bf16_t* __restrict__ gru_bf,         // [S, H]
    unsigned int* __restrict__ syncc, int S) {
  __shared__ float sh[HDIM];
  const int tid = threadIdx.x;
  const int blk = blockIdx.x;
  const int o = tid >> 4, t16 = tid & 15;
  const int i = blk * (HDIM / GRU_WGS) + o;
  const bf16_t* wr = Whht + (size_t)i * HDIM;
  const bf16_t* wz = Whht + (size_t)(i + HDIM) * HDIM;
  const bf16_t* wn = Whht + (size_t)(i + 2 * HDIM) * HDIM;
  const float br = b_hh[i], bz = b_hh[i + HDIM], bn = b_hh[i + 2 * HDIM];
  const int kb = t16 * 64;

  for (int t = 0; t < S; ++t) {
    const float* hc = hbuf + (t & 1) * HDIM;
    float* hn = hbuf + ((t + 1) & 1) * HDIM;
    sh[tid] = hc[tid];
    __syncthreads();

    float pr = 0.f, pz = 0.f, pn = 0.f;
#pragma unroll
    for (int kk = 0; kk < 64; kk += 8) {
      bf16x8 wrv = *(const bf16x8*)(wr + kb + kk);
      bf16x8 wzv = *(const bf16x8*)(wz + kb + kk);
      bf16x8 wnv = *(const bf16x8*)(wn + kb + kk);
#pragma unroll
      for (int j = 0; j < 8; ++j) {
        float hv = sh[kb + kk + j];
        pr = fmaf(hv, (float)wrv[j], pr);
        pz = fmaf(hv, (float)wzv[j], pz);
        pn = fmaf(hv, (float)wnv[j], pn);
      }
    }
    for (int d = 8; d > 0; d >>= 1) {
      pr += __shfl_down(pr, d, 16);
      pz += __shfl_down(pz, d, 16);
      pn += __shfl_down(pn, d, 16);
    }
    if (t16 == 0) {
      const float* xt = xw + (size_t)t * 3 * HDIM;
      float r = 1.f / (1.f + __expf(-(xt[i] + pr + br)));
      float z = 1.f / (1.f + __expf(-(xt[i + HDIM] + pz + bz)));
      float n = tanhf(xt[i + 2 * HDIM] + r * (pn + bn));
      float hv = (1.f - z) * n + z * sh[i];
      hn[i] = hv;
      gru_bf[(size_t)t * HDIM + i] = (bf16_t)hv;
    }
    __threadfence();
    __syncthreads();
    __builtin_amdgcn_s_cluster_barrier();       // NOP if not in a cluster
    if (tid == 0) {
      atomicAdd(syncc, 1u);
      volatile unsigned int* vs = (volatile unsigned int*)syncc;
      unsigned int tgt = (unsigned int)(t + 1) * GRU_WGS;
      while (*vs < tgt) { __builtin_amdgcn_s_sleep(1); }
      __threadfence();
    }
    __syncthreads();
  }
}

__global__ void k_init(float* __restrict__ hbuf, const float* __restrict__ h0,
                       unsigned int* __restrict__ syncc) {
  int t = threadIdx.x;
  hbuf[t] = h0[t];
  if (t == 0) *syncc = 0u;
}

__global__ void k_copy_h(float* __restrict__ dst, const float* __restrict__ src) {
  dst[threadIdx.x] = src[threadIdx.x];
}

// ===========================================================================
extern "C" void kernel_launch(void* const* d_in, const int* in_sizes, int n_in,
                              void* d_out, int out_size, void* d_ws, size_t ws_size,
                              hipStream_t stream) {
  (void)in_sizes; (void)n_in; (void)out_size; (void)ws_size;
  const int S = S_LEN;

  const float* inp_seq = (const float*)d_in[0];
  const float* h0      = (const float*)d_in[1];
  const float* h_start = (const float*)d_in[2];
  const float* bb      = (const float*)d_in[3];
  const float* gt      = (const float*)d_in[4];
  const int*   hier    = (const int*)d_in[69];
  auto F = [&](int i) { return (const float*)d_in[i]; };

  float* out_main = (float*)d_out;                      // [S,63]
  float* out_next = out_main + (size_t)S * 63;          // [S,1024]
  float* out_leaf = out_next + (size_t)S * HDIM;        // [S,1]
  float* out_h    = out_leaf + S;                       // [1024]

  size_t off = 0;
  auto alloc = [&](size_t bytes) -> void* {
    void* p = (char*)d_ws + off;
    off += (bytes + 255) & ~(size_t)255;
    return p;
  };

  bf16_t* x0bf  = (bf16_t*)alloc((size_t)S * 128 * 2);
  bf16_t* t1    = (bf16_t*)alloc((size_t)S * HDIM * 2);
  bf16_t* t2    = (bf16_t*)alloc((size_t)S * HDIM * 2);
  bf16_t* xbf   = (bf16_t*)alloc((size_t)S * HDIM * 2);
  float*  xw    = (float*) alloc((size_t)S * 3 * HDIM * 4);
  bf16_t* grubf = (bf16_t*)alloc((size_t)S * HDIM * 2);
  float*  hbuf  = (float*) alloc(2 * HDIM * 4);
  unsigned int* syncc = (unsigned int*)alloc(256);
  bf16_t* cbuf  = (bf16_t*)alloc((size_t)S * 2048 * 2);
  bf16_t* hA    = (bf16_t*)alloc((size_t)S * HDIM * 2);
  float*  hB    = (float*) alloc((size_t)S * HDIM * 4);
  bf16_t* nA    = (bf16_t*)alloc((size_t)S * HDIM * 2);
  bf16_t* nB    = (bf16_t*)alloc((size_t)S * HDIM * 2);

  auto cv = [&](const float* Wp, int K, int N, int Kpad) -> bf16_t* {
    bf16_t* d = (bf16_t*)alloc((size_t)N * Kpad * 2);
    long long tot = (long long)N * Kpad;
    k_convert_w<<<(int)((tot + 255) / 256), 256, 0, stream>>>(Wp, d, K, N, Kpad);
    return d;
  };

  // All Kpad multiples of 64 (TDM chunk size)
  bf16_t* inpW1t  = cv(F(5),  69,   1024, 128);
  bf16_t* inpW2t  = cv(F(7),  1024, 1024, 1024);
  bf16_t* inpW3t  = cv(F(9),  1024, 1024, 1024);
  bf16_t* cmdW1t  = cv(F(11), 1024, 512,  1024);
  bf16_t* cmdW2t  = cv(F(13), 512,  256,  512);
  bf16_t* cubeW1t = cv(F(17), 1031, 512,  1088);
  bf16_t* cubeW2t = cv(F(19), 512,  256,  512);
  bf16_t* dimW1t  = cv(F(23), 1027, 512,  1088);
  bf16_t* dimW2t  = cv(F(25), 512,  256,  512);
  bf16_t* alnW1t  = cv(F(29), 1027, 512,  1088);
  bf16_t* alnW2t  = cv(F(31), 512,  256,  512);
  bf16_t* attW1t  = cv(F(35), 1046, 1024, 1088);
  bf16_t* attW2t  = cv(F(37), 1024, 1024, 1024);
  bf16_t* symW1t  = cv(F(41), 1038, 512,  1088);
  bf16_t* symW2t  = cv(F(43), 512,  256,  512);
  bf16_t* sqzW1t  = cv(F(47), 1057, 512,  1088);
  bf16_t* sqzW2t  = cv(F(49), 512,  256,  512);
  bf16_t* nxtW1t  = cv(F(53), 2048, 1024, 2048);
  bf16_t* nxtW2t  = cv(F(55), 1024, 1024, 1024);
  bf16_t* nxtW3t  = cv(F(57), 1024, 1024, 1024);
  bf16_t* lfW1t   = cv(F(59), 2048, 512,  2048);
  bf16_t* lfW2t   = cv(F(61), 512,  128,  512);
  bf16_t* WihT    = cv(F(65), 1024, 3072, 1024);
  bf16_t* WhhT    = cv(F(66), 1024, 3072, 1024);

  auto gemm = [&](const bf16_t* A, const bf16_t* Bt, const float* bias,
                  float* Cf, bf16_t* Cb, int M, int N, int K, int act) {
    dim3 g(N / 64, M / 256);
    k_gemm<<<g, 256, 0, stream>>>(A, Bt, bias, Cf, Cb, M, N, K, act);
  };
  auto concat = [&](int Kpad, const float* extra, int eoff, int ew,
                    const float* bcast, int bw) {
    long long tot = (long long)S * Kpad;
    k_concat<<<(int)((tot + 255) / 256), 256, 0, stream>>>(
        cbuf, Kpad, grubf, HDIM, extra, 63, eoff, ew, bcast, bw, S);
  };
  auto head_out = [&](const float* A, int K, int pbase, float* outp,
                      int ostride, int ooff, int Nout) {
    long long tot = (long long)S * Nout;
    k_head_out<<<(int)((tot + 255) / 256), 256, 0, stream>>>(
        A, K, F(pbase + 4), F(pbase + 5), outp, ostride, ooff, Nout, S);
  };
  auto run_head = [&](const bf16_t* Ain, int Kin, bf16_t* W1t, bf16_t* W2t,
                      int pbase, int N1, int N2, int Nout, int ooff) {
    gemm(Ain, W1t, F(pbase + 1), nullptr, hA, S, N1, Kin, 1);
    gemm(hA,  W2t, F(pbase + 3), hB, nullptr, S, N2, N1, 1);
    head_out(hB, N2, pbase, out_main, 63, ooff, Nout);
  };

  // ---- input MLP + GRU input projection
  {
    long long tot = (long long)S * 128;
    k_build_x0<<<(int)((tot + 255) / 256), 256, 0, stream>>>(inp_seq, bb, hier, x0bf, S);
  }
  gemm(x0bf, inpW1t, F(6),  nullptr, t1,  S, 1024, 128,  1);
  gemm(t1,   inpW2t, F(8),  nullptr, t2,  S, 1024, 1024, 1);
  gemm(t2,   inpW3t, F(10), nullptr, xbf, S, 1024, 1024, 0);
  gemm(xbf,  WihT,   F(67), xw, nullptr,  S, 3072, 1024, 0);   // xw = x@W_ih + b_ih

  // ---- GRU recurrence
  k_init<<<1, 1024, 0, stream>>>(hbuf, h0, syncc);
  k_gru_scan<<<GRU_WGS, 1024, 0, stream>>>(xw, WhhT, F(68), hbuf, grubf, syncc, S);
  k_copy_h<<<1, 1024, 0, stream>>>(out_h, hbuf);   // S even -> final h in buffer 0

  // ---- output heads (out cols: cmd 0, cube 7, dim 40, att 43, sym 49,
  //      squeeze 54, align 62)
  run_head(grubf, 1024, cmdW1t, cmdW2t, 11, 512, 256, 7, 0);

  concat(1088, gt, 0, 7, nullptr, 0);
  run_head(cbuf, 1088, cubeW1t, cubeW2t, 17, 512, 256, 33, 7);

  concat(1088, nullptr, 0, 0, bb, 3);
  run_head(cbuf, 1088, dimW1t, dimW2t, 23, 512, 256, 3, 40);
  run_head(cbuf, 1088, alnW1t, alnW2t, 29, 512, 256, 1, 62);

  concat(1088, gt, 7, 22, nullptr, 0);
  run_head(cbuf, 1088, attW1t, attW2t, 35, 1024, 1024, 6, 43);

  concat(1088, gt, 7, 11, bb, 3);
  run_head(cbuf, 1088, symW1t, symW2t, 41, 512, 256, 5, 49);

  concat(1088, gt, 7, 33, nullptr, 0);
  run_head(cbuf, 1088, sqzW1t, sqzW2t, 47, 512, 256, 8, 54);

  // ---- next_net / leaf_net on [gru | h_start]
  concat(2048, nullptr, 0, 0, h_start, 1024);
  gemm(cbuf, nxtW1t, F(54), nullptr, nA, S, 1024, 2048, 1);
  gemm(nA,   nxtW2t, F(56), nullptr, nB, S, 1024, 1024, 1);
  gemm(nB,   nxtW3t, F(58), out_next, nullptr, S, 1024, 1024, 0);

  gemm(cbuf, lfW1t, F(60), nullptr, hA, S, 512, 2048, 1);
  gemm(hA,   lfW2t, F(62), hB, nullptr, S, 128, 512, 1);
  head_out(hB, 128, 59, out_leaf, 1, 0, 1);
}
